// CBL_1632087573343
// MI455X (gfx1250) — matrix operand order repaired and checked by reading, not compiled, not run on hardware
//
#include <hip/hip_runtime.h>
#include <hip/hip_bf16.h>

typedef __attribute__((ext_vector_type(16))) __bf16 v16bf;
typedef __attribute__((ext_vector_type(8)))  __bf16 v8bf;
typedef __attribute__((ext_vector_type(8)))  float  v8f;

#define B_  8
#define C_  256
#define H_  128
#define W_  128
#define HW_ (H_ * W_)
#define NPIX (B_ * HW_)
#define FEAT_BYTES ((size_t)NPIX * C_ * 2)   // 67,108,864

union BfPack { v16bf v; v8bf h[2]; };

__device__ __forceinline__ size_t pix_idx(int b, int y, int x) {
    return (size_t)((b * H_ + y) * W_ + x);
}

// ---------------------------------------------------------------------------
// Pass 0: zero the small accumulators (must run every launch; harness does not
// re-poison between graph replays).
// ---------------------------------------------------------------------------
__global__ void zero_kernel(float* sum, int* cnt, int* inc) {
    int t = threadIdx.x;
    if (t < B_) { sum[t] = 0.0f; cnt[t] = 0; inc[t] = 0; }
}

// ---------------------------------------------------------------------------
// Pass 1: coalesced sweep of er_input [b,c,h,w]; per-pixel inverse L2 norm;
// write unit vectors as bf16 in pixel-major [b,h,w,c]; per-image cnt/include.
// One block = 32 pixels x 256 channels staged through LDS.
// ---------------------------------------------------------------------------
__global__ void normalize_kernel(const float* __restrict__ er,
                                 const int* __restrict__ seg,
                                 const int* __restrict__ gtb,
                                 __bf16* __restrict__ feat,
                                 int* cnt, int* inc) {
    __shared__ float tile[32][C_ + 1];
    __shared__ float red[32][8];
    __shared__ float rnorm[32];

    const int t = threadIdx.x;
    const size_t pixbase = (size_t)blockIdx.x * 32;

    // load 32 pixels x 256 channels; consecutive threads read consecutive x
    for (int i = t; i < 32 * C_; i += 256) {
        int ch = i >> 5, px = i & 31;
        size_t p = pixbase + px;
        int b = (int)(p >> 14);
        int rem = (int)(p & (HW_ - 1));
        tile[px][ch] = er[((size_t)(b * C_ + ch) << 14) + rem];
    }
    __syncthreads();

    // 8 threads per pixel: partial sum of squares over 32 channels each
    {
        int px = t >> 3, part = t & 7;
        float s = 0.0f;
        #pragma unroll
        for (int k = 0; k < 32; ++k) {
            float v = tile[px][part * 32 + k];
            s += v * v;
        }
        red[px][part] = s;
    }
    __syncthreads();

    if (t < 32) {
        float s = 0.0f;
        #pragma unroll
        for (int k = 0; k < 8; ++k) s += red[t][k];
        float n = fmaxf(sqrtf(s), 1e-8f);
        rnorm[t] = 1.0f / n;

        // per-image masks for this pixel
        size_t p = pixbase + t;
        int b = (int)(p >> 14);
        int rem = (int)(p & (HW_ - 1));
        int y = rem >> 7, x = rem & 127;
        int sv = seg[p], gv = gtb[p];
        int s0 = (sv == 255) ? 0 : sv;
        int g0 = (gv == 255) ? 0 : gv;
        if (s0 * g0 > 0) {
            atomicOr(&inc[b], 1);
            if (y >= 2 && y <= H_ - 3 && x >= 2 && x <= W_ - 3)
                atomicAdd(&cnt[b], 1);
        }
    }
    __syncthreads();

    // write unit vectors, pixel-major; consecutive threads -> consecutive ch
    for (int i = t; i < 32 * C_; i += 256) {
        int px = i >> 8, ch = i & (C_ - 1);
        feat[(pixbase + px) * C_ + ch] = (__bf16)(tile[px][ch] * rnorm[px]);
    }
}

// ---------------------------------------------------------------------------
// Pass 2: one wave32 per (b, y, 16-pixel group, dy). Gram of the 16 A-pixels
// against x0-2..x0+17 of row y+dy via two 16x16 bf16 WMMA accumulator tiles
// (K = 256, 8 steps of 32). Banded entries |dx|<=2 are the needed cosines.
// ---------------------------------------------------------------------------
__global__ void gram_kernel(const __bf16* __restrict__ F,
                            const int* __restrict__ seg,
                            const int* __restrict__ gtb,
                            float* __restrict__ sum) {
    int id = blockIdx.x;
    const int dyi = id % 5;  id /= 5;
    const int xg  = id & 7;  id >>= 3;
    const int y   = id & 127; id >>= 7;
    const int b   = id;                    // 0..7
    if (y < 2 || y > H_ - 3) return;       // no interior-valid A pixel in row

    const int dy = dyi - 2;
    const int yq = y + dy;                 // in [0,127] given y in [2,125]
    const int x0 = xg * 16;

    const int l    = threadIdx.x;
    const int half = l >> 4;               // K-half owned by this lane
    const int N    = l & 15;

    // A: lane owns row (pixel x0+N); B: lane owns column (pixel base+N)
    const __bf16* rowA  = F + pix_idx(b, y, x0 + N) * C_;
    const int xq0c = min(max(x0 - 2 + N, 0), W_ - 1);   // clamped loads feed
    const int xq1c = min(max(x0 + 14 + N, 0), W_ - 1);  // only masked entries
    const __bf16* rowB0 = F + pix_idx(b, yq, xq0c) * C_;
    const __bf16* rowB1 = F + pix_idx(b, yq, xq1c) * C_;

    v8f acc0 = {}, acc1 = {};
    #pragma unroll
    for (int kb = 0; kb < C_; kb += 32) {
        BfPack a, b0, b1;
        // A as loaded here, contiguous k order (matches B): lanes 0-15 K=0..15, lanes 16-31 K=16..31
        a.h[0]  = *(const v8bf*)(rowA  + kb + (half << 4));
        a.h[1]  = *(const v8bf*)(rowA  + kb + (half << 4) + 8);
        // B as loaded here, contiguous k order (matches A): lanes 0-15 K=0..15, lanes 16-31 K=16..31
        b0.h[0] = *(const v8bf*)(rowB0 + kb + (half << 4));
        b0.h[1] = *(const v8bf*)(rowB0 + kb + (half << 4) + 8);
        b1.h[0] = *(const v8bf*)(rowB1 + kb + (half << 4));
        b1.h[1] = *(const v8bf*)(rowB1 + kb + (half << 4) + 8);

        acc0 = __builtin_amdgcn_wmma_f32_16x16x32_bf16(false, a.v, false, b0.v,
                                                       (short)0, acc0, false, false);
        acc1 = __builtin_amdgcn_wmma_f32_16x16x32_bf16(false, a.v, false, b1.v,
                                                       (short)0, acc1, false, false);
    }

    // Extract banded entries: C tile layout VGPR r, lane -> M = r + 8*half, N = l&15
    float lsum = 0.0f;
    #pragma unroll
    for (int r = 0; r < 8; ++r) {
        const int M  = r + (half << 3);
        const int xp = x0 + M;
        if (xp < 2 || xp > W_ - 3) continue;           // interior requirement
        const int sv = seg[pix_idx(b, y, xp)];
        const int gv = gtb[pix_idx(b, y, xp)];
        const int s0 = (sv == 255) ? 0 : sv;
        const int g0 = (gv == 255) ? 0 : gv;
        if (s0 * g0 <= 0) continue;                    // not a boundary pixel
        #pragma unroll
        for (int t = 0; t < 2; ++t) {
            const int xq = x0 - 2 + 16 * t + N;
            const int dx = xq - xp;
            if (dx < -2 || dx > 2) continue;
            if (dx == 0 && dy == 0) continue;          // skip center shift
            const int sq = seg[pix_idx(b, yq, xq)];    // xq in [0,127] here
            const float labsim = (sv == sq && (unsigned)sv < 2u) ? 1.0f : 0.0f;
            const float cs = t ? acc1[r] : acc0[r];
            const float d  = cs - labsim;
            lsum += d * d;
        }
    }

    // wave32 reduction then one atomic per wave
    #pragma unroll
    for (int off = 16; off >= 1; off >>= 1)
        lsum += __shfl_xor(lsum, off, 32);
    if (l == 0) atomicAdd(&sum[b], lsum);
}

// ---------------------------------------------------------------------------
// Pass 3: scalar reduction to the loss.
// ---------------------------------------------------------------------------
__global__ void finalize_kernel(const float* __restrict__ sum,
                                const int* __restrict__ cnt,
                                const int* __restrict__ inc,
                                float* __restrict__ out) {
    if (threadIdx.x == 0 && blockIdx.x == 0) {
        int sn = 0;
        for (int b = 0; b < B_; ++b) sn += inc[b] ? 1 : 0;
        float scale = (float)max(sn, 1);
        float loss = 0.0f;
        for (int b = 0; b < B_; ++b)
            if (inc[b])
                loss += (sum[b] / fmaxf((float)cnt[b], 1.0f)) / 24.0f;
        out[0] = loss / scale;
    }
}

extern "C" void kernel_launch(void* const* d_in, const int* in_sizes, int n_in,
                              void* d_out, int out_size, void* d_ws, size_t ws_size,
                              hipStream_t stream) {
    (void)in_sizes; (void)n_in; (void)out_size; (void)ws_size;
    const float* er  = (const float*)d_in[0];
    const int*   seg = (const int*)d_in[1];
    const int*   gtb = (const int*)d_in[2];
    float* out = (float*)d_out;

    __bf16* feat = (__bf16*)d_ws;
    char* aux = (char*)d_ws + FEAT_BYTES;
    float* sum = (float*)aux;            // 8 floats
    int*   cnt = (int*)(aux + 32);       // 8 ints
    int*   inc = (int*)(aux + 64);       // 8 ints

    zero_kernel<<<1, 32, 0, stream>>>(sum, cnt, inc);
    normalize_kernel<<<NPIX / 32, 256, 0, stream>>>(er, seg, gtb, feat, cnt, inc);
    gram_kernel<<<B_ * H_ * 8 * 5, 32, 0, stream>>>(feat, seg, gtb, sum);
    finalize_kernel<<<1, 32, 0, stream>>>(sum, cnt, inc, out);
}
